// LengthRegulator_2929167696555
// MI455X (gfx1250) — matrix operand order repaired and checked by reading, not compiled, not run on hardware
//
#include <hip/hip_runtime.h>
#include <hip/hip_bf16.h>

typedef _Float16 half_t;
typedef __attribute__((ext_vector_type(4)))  _Float16 v4h;
typedef __attribute__((ext_vector_type(8)))  _Float16 v8h;
typedef __attribute__((ext_vector_type(16))) _Float16 v16h;
typedef __attribute__((ext_vector_type(8)))  float    v8f;
typedef __attribute__((ext_vector_type(4)))  int      v4i;

#define BSZ   16
#define TLEN  512
#define DIM   384
#define KW    3
#define KTOT  (KW * DIM)      // 1152
#define NCHUNK (KTOT / 32)    // 36 k-chunks of 32
#define NTILE  (DIM / 16)     // 24 n-tiles of 16
#define LN_EPS 1e-5f

// gfx1250 async global->LDS path (ASYNCcnt-tracked DMA into LDS, no VGPR transit).
// Guarded: falls back to load+ds_store staging if the toolchain lacks the builtins.
#if __has_builtin(__builtin_amdgcn_global_load_async_to_lds_b128) && \
    __has_builtin(__builtin_amdgcn_s_wait_asynccnt)
#define USE_ASYNC_LDS 1
#else
#define USE_ASYNC_LDS 0
#endif

// ---------------------------------------------------------------------------
// Repack torch-layout conv weights w[Cout, Cin, K] (f32) into f16 WMMA
// B-fragments: wp[((c*NTILE + nt)*32 + lane)*16 + e]
//   lane: n = nt*16 + (lane&15); K-within-chunk = e + (lane>=16 ? 16 : 0)
// (contiguous k order per lane half; the A fragments use the same order)
// ---------------------------------------------------------------------------
__global__ void repack_w_kernel(const float* __restrict__ w, half_t* __restrict__ wp) {
    const int blk  = blockIdx.x;            // NCHUNK*NTILE blocks
    const int c    = blk / NTILE;
    const int nt   = blk % NTILE;
    const int tid  = threadIdx.x;           // 512 threads
    const int lane = tid & 31;
    const int e    = tid >> 5;              // 0..15
    const int K    = e + ((lane & 16) ? 16 : 0);
    const int k    = c * 32 + K;            // global K index = dk*DIM + cin
    const int cin  = k % DIM;
    const int dk   = k / DIM;
    const int n    = nt * 16 + (lane & 15);
    const float v  = w[((size_t)n * DIM + cin) * KW + dk];
    wp[(((size_t)(c * NTILE + nt) * 32) + lane) * 16 + e] = (half_t)v;
}

// ---------------------------------------------------------------------------
// Fused conv1d(K=3, same) -> +bias -> LayerNorm -> ReLU.
// One workgroup computes 16 consecutive rows of (B*T) x 384 output.
// 8 waves x 3 n-tiles; K-loop of 36 chunks of 32 with v_wmma_f32_16x16x32_f16.
// FUSE_PRED: also computes pred = relu_out @ wl + bl (h not stored).
// ---------------------------------------------------------------------------
template <bool IN_HALF, bool FUSE_PRED>
__global__ void conv_ln_relu_kernel(const void* __restrict__ in_v,
                                    const half_t* __restrict__ wp,
                                    const float* __restrict__ bias,
                                    const float* __restrict__ gamma,
                                    const float* __restrict__ beta,
                                    half_t* __restrict__ hout,
                                    const float* __restrict__ wl,
                                    const float* __restrict__ bl,
                                    float* __restrict__ pred) {
    __shared__ __align__(16) half_t xs[18][DIM];     // input panel (halo rows), f16
    __shared__ float part_s1[8][16];
    __shared__ float part_s2[8][16];
    __shared__ float part_p [8][16];
    __shared__ float red_s1[16];
    __shared__ float red_s2[16];

    const int tile    = blockIdx.x;                  // 0..511
    const int batch   = tile / (TLEN / 16);
    const int tbase   = (tile % (TLEN / 16)) * 16;
    const int rowbase = tile * 16;                   // == batch*TLEN + tbase
    const int tid     = threadIdx.x;
    const int lane    = tid & 31;
    const int wv      = tid >> 5;                    // wave 0..7
    const int ncol    = lane & 15;
    const int hi      = (lane >> 4) & 1;
    const int base_k  = hi * 16;

    // --- stage rows tbase-1 .. tbase+16 into LDS as f16 (zero halo) ---
    if constexpr (IN_HALF) {
#if USE_ASYNC_LDS
        // DMA 16-byte chunks straight into LDS (18 rows x 48 chunks).
        for (int idx = tid; idx < 18 * 48; idx += 256) {
            const int r  = idx / 48;
            const int c8 = (idx % 48) * 8;
            const int t  = tbase - 1 + r;
            if (t >= 0 && t < TLEN) {
                const half_t* gp = (const half_t*)in_v + ((size_t)(batch * TLEN + t)) * DIM + c8;
                __builtin_amdgcn_global_load_async_to_lds_b128(
                    (__attribute__((address_space(1))) v4i*)(half_t*)gp,
                    (__attribute__((address_space(3))) v4i*)&xs[r][c8], 0, 0);
            } else {
                v8h z = {};
                *(v8h*)&xs[r][c8] = z;               // halo zero-fill (disjoint addrs)
            }
        }
        __builtin_amdgcn_s_wait_asynccnt(0);
#else
        for (int idx = tid; idx < 18 * 96; idx += 256) {
            const int r  = idx / 96;
            const int c4 = (idx % 96) * 4;
            const int t  = tbase - 1 + r;
            v4h hv;
            if (t >= 0 && t < TLEN) {
                const half_t* p = (const half_t*)in_v + ((size_t)(batch * TLEN + t)) * DIM + c4;
                hv = *(const v4h*)p;
            } else {
                hv[0] = (half_t)0.f; hv[1] = (half_t)0.f;
                hv[2] = (half_t)0.f; hv[3] = (half_t)0.f;
            }
            *(v4h*)&xs[r][c4] = hv;
        }
#endif
    } else {
        for (int idx = tid; idx < 18 * 96; idx += 256) {
            const int r  = idx / 96;
            const int c4 = (idx % 96) * 4;
            const int t  = tbase - 1 + r;
            v4h hv;
            if (t >= 0 && t < TLEN) {
                const float4 f = *(const float4*)((const float*)in_v +
                                   ((size_t)(batch * TLEN + t)) * DIM + c4);
                hv[0] = (half_t)f.x; hv[1] = (half_t)f.y;
                hv[2] = (half_t)f.z; hv[3] = (half_t)f.w;
            } else {
                hv[0] = (half_t)0.f; hv[1] = (half_t)0.f;
                hv[2] = (half_t)0.f; hv[3] = (half_t)0.f;
            }
            *(v4h*)&xs[r][c4] = hv;
        }
    }
    __syncthreads();

    const int m   = ncol;                            // A-row handled by this lane
    const int nt0 = wv * 3;                          // first of 3 n-tiles

    v8f acc0 = {}, acc1 = {}, acc2 = {};
    union AF { v16h v; v8h h[2]; };

    // --- K loop: 36 chunks of 32 (chunk never straddles a dk boundary) ---
    for (int c = 0; c < NCHUNK; ++c) {
        const int dk   = c / 12;
        const int cinb = (c % 12) * 32;

        AF af;  // A layout: lane half h holds K=16h+0..15 contiguous (matches B repack)
        af.h[0] = *(const v8h*)&xs[m + dk][cinb + base_k];
        af.h[1] = *(const v8h*)&xs[m + dk][cinb + base_k + 8];

        const size_t bidx = (((size_t)(c * NTILE + nt0) * 32) + lane) * 16;
        const v16h bf0 = *(const v16h*)(wp + bidx);
        const v16h bf1 = *(const v16h*)(wp + bidx + 32 * 16);
        const v16h bf2 = *(const v16h*)(wp + bidx + 2 * 32 * 16);

        acc0 = __builtin_amdgcn_wmma_f32_16x16x32_f16(false, af.v, false, bf0, (short)0, acc0, false, false);
        acc1 = __builtin_amdgcn_wmma_f32_16x16x32_f16(false, af.v, false, bf1, (short)0, acc1, false, false);
        acc2 = __builtin_amdgcn_wmma_f32_16x16x32_f16(false, af.v, false, bf2, (short)0, acc2, false, false);
    }

    // --- bias ---
    const int n0 = (nt0 + 0) * 16 + ncol;
    const int n1 = (nt0 + 1) * 16 + ncol;
    const int n2 = (nt0 + 2) * 16 + ncol;
    const float bia0 = bias[n0], bia1 = bias[n1], bia2 = bias[n2];
#pragma unroll
    for (int r = 0; r < 8; ++r) { acc0[r] += bia0; acc1[r] += bia1; acc2[r] += bia2; }

    // --- LayerNorm stats: per-row sum / sumsq, deterministic reduction ---
#pragma unroll
    for (int r = 0; r < 8; ++r) {
        const float v0 = acc0[r], v1 = acc1[r], v2 = acc2[r];
        float s1 = v0 + v1 + v2;
        float s2 = v0 * v0 + v1 * v1 + v2 * v2;
#pragma unroll
        for (int msk = 1; msk < 16; msk <<= 1) {
            s1 += __shfl_xor(s1, msk);
            s2 += __shfl_xor(s2, msk);
        }
        if (ncol == 0) { part_s1[wv][r + hi * 8] = s1; part_s2[wv][r + hi * 8] = s2; }
    }
    __syncthreads();
    if (tid < 16) {
        float a = 0.f, s = 0.f;
        for (int w = 0; w < 8; ++w) { a += part_s1[w][tid]; s += part_s2[w][tid]; }
        red_s1[tid] = a; red_s2[tid] = s;
    }
    __syncthreads();

    // --- normalize, ReLU, emit ---
    const float g0 = gamma[n0], g1 = gamma[n1], g2 = gamma[n2];
    const float e0 = beta[n0],  e1 = beta[n1],  e2 = beta[n2];
    float wl0 = 0.f, wl1 = 0.f, wl2 = 0.f;
    if constexpr (FUSE_PRED) { wl0 = wl[n0]; wl1 = wl[n1]; wl2 = wl[n2]; }

#pragma unroll
    for (int r = 0; r < 8; ++r) {
        const int mrow = r + hi * 8;
        const float mean = red_s1[mrow] * (1.0f / DIM);
        const float var  = red_s2[mrow] * (1.0f / DIM) - mean * mean;
        const float inv  = rsqrtf(var + LN_EPS);
        const float y0 = fmaxf((acc0[r] - mean) * inv * g0 + e0, 0.f);
        const float y1 = fmaxf((acc1[r] - mean) * inv * g1 + e1, 0.f);
        const float y2 = fmaxf((acc2[r] - mean) * inv * g2 + e2, 0.f);
        if constexpr (!FUSE_PRED) {
            half_t* hp = hout + (size_t)(rowbase + mrow) * DIM;
            hp[n0] = (half_t)y0; hp[n1] = (half_t)y1; hp[n2] = (half_t)y2;
        }
        if constexpr (FUSE_PRED) {
            float p = y0 * wl0 + y1 * wl1 + y2 * wl2;
#pragma unroll
            for (int msk = 1; msk < 16; msk <<= 1) p += __shfl_xor(p, msk);
            if (ncol == 0) part_p[wv][mrow] = p;
        }
    }
    if constexpr (FUSE_PRED) {
        __syncthreads();
        if (tid < 16) {
            float p = bl[0];
            for (int w = 0; w < 8; ++w) p += part_p[w][tid];
            pred[rowbase + tid] = p;
        }
    }
}

// ---------------------------------------------------------------------------
// Per-batch inclusive cumsum of durations (T=512, trivial).
// ---------------------------------------------------------------------------
__global__ void cumsum_kernel(const int* __restrict__ dur, int* __restrict__ cum) {
    const int b = blockIdx.x;
    if (threadIdx.x == 0) {
        int s = 0;
        for (int j = 0; j < TLEN; ++j) { s += dur[b * TLEN + j]; cum[b * TLEN + j] = s; }
    }
}

// ---------------------------------------------------------------------------
// expanded[b,l,:] = x[b, j(l), :] where cum[j-1] <= l < cum[j]; zeros past total.
// ---------------------------------------------------------------------------
__global__ void expand_kernel(const float* __restrict__ x, const int* __restrict__ cum,
                              float* __restrict__ out, int L) {
    __shared__ int cums[TLEN];
    const int b = blockIdx.y;
    for (int i = threadIdx.x; i < TLEN; i += 256) cums[i] = cum[b * TLEN + i];
    __syncthreads();
    const int total = cums[TLEN - 1];

    const int row   = threadIdx.x >> 6;   // 4 rows per block
    const int lanec = threadIdx.x & 63;
    const int l = blockIdx.x * 4 + row;
    if (l >= L) return;

    float* orow = out + ((size_t)b * L + l) * DIM;
    if (l >= total) {
        for (int c = lanec; c < DIM; c += 64) orow[c] = 0.f;
    } else {
        int lo = 0, hicur = TLEN - 1;     // smallest j with cums[j] > l
        while (lo < hicur) {
            const int mid = (lo + hicur) >> 1;
            if (cums[mid] > l) hicur = mid; else lo = mid + 1;
        }
        const float* xrow = x + ((size_t)b * TLEN + lo) * DIM;
        for (int c = lanec; c < DIM; c += 64) orow[c] = xrow[c];
    }
}

// ---------------------------------------------------------------------------
extern "C" void kernel_launch(void* const* d_in, const int* in_sizes, int n_in,
                              void* d_out, int out_size, void* d_ws, size_t ws_size,
                              hipStream_t stream) {
    const float* x   = (const float*)d_in[0];
    const int*   dur = (const int*)d_in[1];
    const float* w1  = (const float*)d_in[2];
    const float* b1  = (const float*)d_in[3];
    const float* g1  = (const float*)d_in[4];
    const float* be1 = (const float*)d_in[5];
    const float* w2  = (const float*)d_in[6];
    const float* b2  = (const float*)d_in[7];
    const float* g2  = (const float*)d_in[8];
    const float* be2 = (const float*)d_in[9];
    const float* wl  = (const float*)d_in[10];
    const float* bl  = (const float*)d_in[11];

    const int L = (out_size - BSZ * TLEN) / (BSZ * DIM);
    float* expanded = (float*)d_out;
    float* pred     = (float*)d_out + (size_t)BSZ * L * DIM;

    // workspace layout (256B-aligned chunks)
    char* ws = (char*)d_ws;
    const size_t WPACK = (size_t)NCHUNK * NTILE * 32 * 16 * sizeof(half_t); // 884736
    half_t* w1p = (half_t*)ws;
    half_t* w2p = (half_t*)(ws + WPACK);
    half_t* h1  = (half_t*)(ws + 2 * WPACK);
    int*    cum = (int*)(ws + 2 * WPACK + (size_t)BSZ * TLEN * DIM * sizeof(half_t));

    repack_w_kernel<<<NCHUNK * NTILE, 512, 0, stream>>>(w1, w1p);
    repack_w_kernel<<<NCHUNK * NTILE, 512, 0, stream>>>(w2, w2p);

    conv_ln_relu_kernel<false, false><<<(BSZ * TLEN) / 16, 256, 0, stream>>>(
        (const void*)x, w1p, b1, g1, be1, h1, nullptr, nullptr, nullptr);
    conv_ln_relu_kernel<true, true><<<(BSZ * TLEN) / 16, 256, 0, stream>>>(
        (const void*)h1, w2p, b2, g2, be2, nullptr, wl, bl, pred);

    cumsum_kernel<<<BSZ, 32, 0, stream>>>(dur, cum);
    if (L > 0) {
        dim3 grid((L + 3) / 4, BSZ);
        expand_kernel<<<grid, 256, 0, stream>>>(x, cum, expanded, L);
    }
}
